// TD3ActorDSNN_49185965474305
// MI455X (gfx1250) — compile-verified, matrix operand order checked by reading, not run on hardware
//
#include <hip/hip_runtime.h>

// CDNA5 / gfx1250, wave32. WMMA f16 16x16x32, f32 accumulate.

typedef __attribute__((ext_vector_type(16))) _Float16 v16h;
typedef __attribute__((ext_vector_type(8)))  float    v8f;

#define SNN_ALPHA  0.9f
#define SNN_BETA   0.85f
#define SNN_THRESH 1.0f
#define SNN_STEPS  15          // SIM_TIME - 1
#define B_TOTAL    4096
#define D_IN       512
#define HID        2048
#define D_OUT      8

// Scheduler directive masks (LLVM AMDGPU sched_group_barrier)
#define SG_WMMA    0x008
#define SG_VMEM_RD 0x020
#define SG_DS_RD   0x100

// ---------------------------------------------------------------------------
// Fragment addressing for 16-bit WMMA operands (16x32 A tile / 32x16 B tile).
// Per ISA 7.12.2: lanes 0-15 hold K pairs {0..7,16..23}; lanes 16-31 hold
// {8..15,24..31}; each lane stores its 16 halves contiguously (32B) in our
// packing so the hot loop issues contiguous b128 loads.
// ---------------------------------------------------------------------------
__device__ __forceinline__ int frag_half_idx(int mn_local, int k_local) {
  int lane = mn_local + (((k_local & 15) >= 8) ? 16 : 0);
  int j    = ((k_local >= 16) ? 8 : 0) + (k_local & 7);
  return lane * 16 + j;
}

__device__ __forceinline__ v8f wmma_f16(v16h a, v16h b, v8f c) {
  return __builtin_amdgcn_wmma_f32_16x16x32_f16(
      /*neg_a=*/false, a, /*neg_b=*/false, b,
      /*c_mod=*/(short)0, c, /*reuse_a=*/false, /*reuse_b=*/false);
}

// Enforce per-iteration pipeline: A (DS) loads, then ALL B (VMEM) loads as one
// clause, then the 8 WMMAs. This lets the waitcnt pass emit descending partial
// s_wait_loadcnt values so one L2 latency is amortized over 8 WMMAs.
__device__ __forceinline__ void sched_tile_pipeline() {
  __builtin_amdgcn_sched_group_barrier(SG_DS_RD,   2, 0);   // 2x ds_load_b128
  __builtin_amdgcn_sched_group_barrier(SG_VMEM_RD, 16, 0);  // 16x global_load_b128
  __builtin_amdgcn_sched_group_barrier(SG_WMMA,    8, 0);   // 8x v_wmma
}

// ---------------------------------------------------------------------------
// Pack a row-major f32 weight matrix [K][N] into f16 B-fragment order:
// tile (kt, nt) -> 512 halves at ((nt * (K/32) + kt) * 512), lane*16 + j.
// ---------------------------------------------------------------------------
__global__ void pack_w_f16(const float* __restrict__ W, _Float16* __restrict__ Wp,
                           int K, int N) {
  int idx = blockIdx.x * blockDim.x + threadIdx.x;
  if (idx >= K * N) return;
  int k = idx / N, n = idx % N;
  int kt = k >> 5, nt = n >> 4;
  int nkt = K >> 5;
  int base = (nt * nkt + kt) * 512;
  Wp[base + frag_half_idx(n & 15, k & 31)] = (_Float16)W[idx];
}

// ---------------------------------------------------------------------------
// h0 = X @ W0   [4096,512] x [512,2048] -> f32 [4096,2048]
// One WG = 16 output rows x 2048 cols; 512 threads = 16 waves, 8 coltiles/wave.
// ---------------------------------------------------------------------------
__global__ __launch_bounds__(512)
void gemm_h0_kernel(const float* __restrict__ X, const _Float16* __restrict__ W0p,
                    float* __restrict__ H0) {
  __shared__ __align__(32) _Float16 afrag[16 * 512];   // 16 k-tiles of A, 16KB
  const int tid = threadIdx.x;
  const int rowbase = blockIdx.x * 16;

  // Stage input rows as f16 A fragments (thread tid covers column k=tid).
  {
    const int kt = tid >> 5, kl = tid & 31;
#pragma unroll
    for (int i = 0; i < 16; ++i) {
      float v = X[(rowbase + i) * D_IN + tid];
      afrag[kt * 512 + frag_half_idx(i, kl)] = (_Float16)v;
    }
  }
  __syncthreads();

  const int wave = tid >> 5, lane = tid & 31;
  v8f acc[8];
#pragma unroll
  for (int q = 0; q < 8; ++q)
#pragma unroll
    for (int i = 0; i < 8; ++i) acc[q][i] = 0.0f;

  for (int kt = 0; kt < 16; ++kt) {
    v16h a = *(const v16h*)&afrag[kt * 512 + lane * 16];
    v16h b[8];
#pragma unroll
    for (int q = 0; q < 8; ++q)
      b[q] = *(const v16h*)&W0p[((wave * 8 + q) * 16 + kt) * 512 + lane * 16];
#pragma unroll
    for (int q = 0; q < 8; ++q)
      acc[q] = wmma_f16(a, b[q], acc[q]);
    sched_tile_pipeline();
  }

  const int mbase = (lane & 16) ? 8 : 0;
  const int ncol  = lane & 15;
#pragma unroll
  for (int q = 0; q < 8; ++q) {
    int nt = wave * 8 + q;
#pragma unroll
    for (int i = 0; i < 8; ++i)
      H0[(rowbase + mbase + i) * HID + nt * 16 + ncol] = acc[q][i];
  }
}

// ---------------------------------------------------------------------------
// Persistent fused time loop. One WG owns 32 batch rows; the whole 15-step
// recurrence runs in-kernel (rows are independent -> no global sync needed).
// 1024 threads = 32 waves: wave w -> row-tile (w>>4), col-tiles (w&15)*8..+8.
// mem0/syn1/mem1 stream through global ws (all L2-resident); mem2 lives in
// VGPRs of threads 0..255. Spikes round-trip through a 128KB LDS buffer
// (A-fragment layout for the WMMA, then plain layout for the W2 reduction).
// Keeping the elementwise state in ws (not VGPRs) leaves the GEMM loop with a
// small live set (acc 64 + B 64 + A 8) so 8 B-fragments can stay resident.
// ---------------------------------------------------------------------------
__global__ __launch_bounds__(1024)
void dsnn_steps_kernel(const float* __restrict__ H0, const _Float16* __restrict__ W1p,
                       const float* __restrict__ W2, float* __restrict__ Mem0,
                       float* __restrict__ Syn1, float* __restrict__ Mem1,
                       float* __restrict__ Out) {
  __shared__ __align__(32) _Float16 spike_buf[32 * HID];   // 128KB: s0 frags / s1 plain
  __shared__ float hred[32 * 32 * D_OUT];                  // 32KB reduction buffer

  const int tid  = threadIdx.x;
  const int lane = tid & 31;
  const int wave = tid >> 5;
  const int rowbase = blockIdx.x * 32;
  const int rt = wave >> 4;            // 0..1  (GEMM row tile)
  const int ntbase = (wave & 15) * 8;  // 8 column tiles per wave

  float mem2r = 0.0f;                  // output integrator (tid<256: row,o)

  for (int t = 0; t < SNN_STEPS; ++t) {
    // ---- layer 0: nm0 = BETA*mem0 + h0 ; spike ; reset ; emit s0 A-frags ----
#pragma unroll
    for (int e = 0; e < 64; ++e) {
      int m = e >> 1;
      int k = ((e & 1) << 10) + tid;
      size_t gidx = (size_t)(rowbase + m) * HID + k;
      float prev = (t == 0) ? 0.0f : Mem0[gidx];     // never read poisoned ws
      float nm = SNN_BETA * prev + H0[gidx];
      bool sp = nm > SNN_THRESH;
      Mem0[gidx] = sp ? 0.0f : nm;
      int kt = k >> 5;
      spike_buf[(m >> 4) * 32768 + kt * 512 + frag_half_idx(m & 15, k & 31)] =
          (_Float16)(sp ? 1.0f : 0.0f);
    }
    __syncthreads();

    // ---- h1 = s0 @ W1 : 8 col-tiles x 64 k-tiles of WMMA per wave ----
    v8f acc[8];
#pragma unroll
    for (int q = 0; q < 8; ++q)
#pragma unroll
      for (int i = 0; i < 8; ++i) acc[q][i] = 0.0f;

    const _Float16* abase = &spike_buf[rt * 32768];
    for (int kt = 0; kt < 64; ++kt) {
      v16h a = *(const v16h*)&abase[kt * 512 + lane * 16];
      v16h b[8];
#pragma unroll
      for (int q = 0; q < 8; ++q)
        b[q] = *(const v16h*)&W1p[((size_t)((ntbase + q) * 64 + kt)) * 512 + lane * 16];
#pragma unroll
      for (int q = 0; q < 8; ++q)
        acc[q] = wmma_f16(a, b[q], acc[q]);
      sched_tile_pipeline();
    }

    // ---- layer 1 update on the fragment; syn1/mem1 streamed via ws ----
    {
      const int mb_g = rowbase + rt * 16 + ((lane & 16) ? 8 : 0);
      const int nc_g = lane & 15;
#pragma unroll
      for (int q = 0; q < 8; ++q) {
        const int colb = (ntbase + q) * 16 + nc_g;
#pragma unroll
        for (int i = 0; i < 8; ++i) {
          size_t sidx = (size_t)(mb_g + i) * HID + colb;
          float syn_p = (t == 0) ? 0.0f : Syn1[sidx];
          float mem_p = (t == 0) ? 0.0f : Mem1[sidx];
          float ns = SNN_ALPHA * syn_p + acc[q][i];
          float nm = SNN_BETA * mem_p + ns;
          bool sp = nm > SNN_THRESH;
          Syn1[sidx] = ns;
          Mem1[sidx] = sp ? 0.0f : nm;
          acc[q][i] = sp ? 1.0f : 0.0f;   // stash s1 in acc
        }
      }
    }
    __syncthreads();   // all A-fragment reads of spike_buf complete

    // ---- write s1 to LDS in plain [row][col] f16 layout ----
    {
      const int mb = rt * 16 + ((lane & 16) ? 8 : 0);
      const int nc = lane & 15;
#pragma unroll
      for (int q = 0; q < 8; ++q)
#pragma unroll
        for (int i = 0; i < 8; ++i)
          spike_buf[(mb + i) * HID + (ntbase + q) * 16 + nc] = (_Float16)acc[q][i];
    }
    __syncthreads();

    // ---- h2 = s1 @ W2 (K=2048, N=8): 32 partial segments per row ----
    {
      const int row = tid >> 5, seg = tid & 31;
      float a8[D_OUT];
#pragma unroll
      for (int o = 0; o < D_OUT; ++o) a8[o] = 0.0f;
      const int kb = seg * 64;
      for (int kk = 0; kk < 64; ++kk) {
        float s = (float)spike_buf[row * HID + kb + kk];
        if (s > 0.5f) {
          const float* w2r = &W2[(size_t)(kb + kk) * D_OUT];
#pragma unroll
          for (int o = 0; o < D_OUT; ++o) a8[o] += w2r[o];
        }
      }
#pragma unroll
      for (int o = 0; o < D_OUT; ++o)
        hred[(row * 32 + seg) * D_OUT + o] = a8[o];
    }
    __syncthreads();

    // ---- reduce segments, integrate mem2 (non-leaky) ----
    if (tid < 256) {
      const int row = tid >> 3, o = tid & 7;
      float s = 0.0f;
#pragma unroll
      for (int seg = 0; seg < 32; ++seg)
        s += hred[(row * 32 + seg) * D_OUT + o];
      mem2r += s;
    }
    __syncthreads();
  }

  // ---- output: tanh(mem2) ----
  if (tid < 256) {
    const int row = tid >> 3, o = tid & 7;
    Out[(size_t)(rowbase + row) * D_OUT + o] = tanhf(mem2r);
  }
}

// ---------------------------------------------------------------------------
// Launch. Inputs (setup_inputs order): inputs f32[4096,512], W0 f32[512,2048],
// W1 f32[2048,2048], W2 f32[2048,8]. Output f32[4096,8].
// Workspace: [0,2MB)    W0 f16 packed
//            [2,10MB)   W1 f16 packed
//            [10,42MB)  h0 f32
//            [42,74MB)  mem0 f32
//            [74,106MB) syn1 f32
//            [106,138MB) mem1 f32
// ---------------------------------------------------------------------------
extern "C" void kernel_launch(void* const* d_in, const int* in_sizes, int n_in,
                              void* d_out, int out_size, void* d_ws, size_t ws_size,
                              hipStream_t stream) {
  (void)in_sizes; (void)n_in; (void)out_size; (void)ws_size;
  const float* inputs = (const float*)d_in[0];
  const float* W0 = (const float*)d_in[1];
  const float* W1 = (const float*)d_in[2];
  const float* W2 = (const float*)d_in[3];
  float* out = (float*)d_out;

  char* ws = (char*)d_ws;
  _Float16* W0p  = (_Float16*)(ws);
  _Float16* W1p  = (_Float16*)(ws + (size_t)(2u   << 20));
  float*    H0   = (float*)   (ws + (size_t)(10u  << 20));
  float*    Mem0 = (float*)   (ws + (size_t)(42u  << 20));
  float*    Syn1 = (float*)   (ws + (size_t)(74u  << 20));
  float*    Mem1 = (float*)   (ws + (size_t)(106u << 20));

  pack_w_f16<<<(D_IN * HID + 255) / 256, 256, 0, stream>>>(W0, W0p, D_IN, HID);
  pack_w_f16<<<(HID * HID + 255) / 256, 256, 0, stream>>>(W1, W1p, HID, HID);
  gemm_h0_kernel<<<B_TOTAL / 16, 512, 0, stream>>>(inputs, W0p, H0);
  dsnn_steps_kernel<<<B_TOTAL / 32, 1024, 0, stream>>>(H0, W1p, W2, Mem0, Syn1, Mem1, out);
}
